// ExpertGate_54769422958702
// MI455X (gfx1250) — compile-verified, matrix operand order checked by reading, not run on hardware
//
#include <hip/hip_runtime.h>
#include <hip/hip_bf16.h>
#include <math.h>

// ---- problem constants (from reference) ----
#define D_DIM   2048
#define E_DIM   64
#define TOPK    8
#define ROUTE_SCALE 2.5f
#define KT_TILES (D_DIM / 32)            // 64 K-tiles of 32
#define ET_TILES (E_DIM / 16)            // 4  N-tiles of 16
#define TOK_PER_BLOCK 64                 // 4 waves x 16 tokens
#define NTHREADS 128
#define FRAG_ELEMS (32 * 16)             // one B fragment: 32 lanes x 16 bf16 (1 KB)
#define B_FRAGS (KT_TILES * ET_TILES)
#define B_TOTAL (B_FRAGS * FRAG_ELEMS)   // 131072 bf16 = 256 KB per (hi|lo)
#define CHUNK_KT 8                       // K-tiles staged per LDS chunk
#define N_CHUNKS (KT_TILES / CHUNK_KT)   // 8
#define CHUNK_ELEMS (CHUNK_KT * ET_TILES * FRAG_ELEMS) // 16384 bf16 = 32 KB
#define UNITS_PER_THREAD (CHUNK_ELEMS * 2 / 16 / NTHREADS) // 16 x b128 per thread per (hi|lo)

typedef __attribute__((ext_vector_type(16))) __bf16 v16bf;
typedef __attribute__((ext_vector_type(8)))  float  v8f;

// ------------------------------------------------------------------
// Prep: swizzle W (E=64 x D=2048, f32 row-major) into per-wave B
// fragments, split into bf16 hi/lo.  B is the 32x16 (KxN) operand:
// lane l<16 -> N=l,    elements 0..7 = K 0..7,  8..15 = K 16..23
// lane l>=16 -> N=l-16, elements 0..7 = K 8..15, 8..15 = K 24..31
// Fragment storage: [kt][et][lane][16] contiguous -> tid*16.
// ------------------------------------------------------------------
__global__ void prep_w_kernel(const float* __restrict__ W,
                              __bf16* __restrict__ BH,
                              __bf16* __restrict__ BL) {
    int tid  = blockIdx.x * blockDim.x + threadIdx.x;   // 0..8191
    if (tid >= B_FRAGS * 32) return;
    int lane = tid & 31;
    int et   = (tid >> 5) & (ET_TILES - 1);
    int kt   = tid >> 7;
    int n    = lane & 15;
    int half = lane >> 4;
    int e    = et * 16 + n;
    const float* wrow = W + (size_t)e * D_DIM + kt * 32 + half * 8;
    int base = tid * 16;
#pragma unroll
    for (int i = 0; i < 8; ++i) {
        float v0 = wrow[i];        // K = half*8 + i
        float v1 = wrow[16 + i];   // K = 16 + half*8 + i
        __bf16 h0 = (__bf16)v0;
        __bf16 h1 = (__bf16)v1;
        BH[base + i]     = h0;
        BH[base + 8 + i] = h1;
        BL[base + i]     = (__bf16)(v0 - (float)h0);
        BL[base + 8 + i] = (__bf16)(v1 - (float)h1);
    }
}

// Issue one chunk (32 KB) of async global->LDS b128 copies, spread over the block.
__device__ __forceinline__ void async_copy_chunk(const __bf16* __restrict__ g,
                                                 __bf16* __restrict__ s, int tid) {
#pragma unroll
    for (int i = 0; i < UNITS_PER_THREAD; ++i) {
        int u = i * NTHREADS + tid;                       // 16-byte unit index
        unsigned    lds = (unsigned)(uintptr_t)(s + u * 8);
        const void* gp  = (const void*)(g + u * 8);
        asm volatile("global_load_async_to_lds_b128 %0, %1, off"
                     :: "v"(lds), "v"(gp) : "memory");
    }
}

// ------------------------------------------------------------------
// Main: GEMM (split bf16, B staged via async LDS double-buffer)
//       + sigmoid + biased top-8 + renormalize.
// ------------------------------------------------------------------
__global__ __launch_bounds__(NTHREADS)
void expert_gate_kernel(const float* __restrict__ X,
                        const __bf16* __restrict__ BH,
                        const __bf16* __restrict__ BL,
                        const float* __restrict__ bias,
                        float* __restrict__ out_w,
                        int*   __restrict__ out_i) {
    __shared__ __bf16 s_bh[2][CHUNK_ELEMS];               // 2 x 32 KB
    __shared__ __bf16 s_bl[2][CHUNK_ELEMS];               // 2 x 32 KB
    __shared__ float  s_scores[TOK_PER_BLOCK][E_DIM + 1]; // pitch 65: conflict-free columns
    __shared__ float  s_bias[E_DIM];

    const int tid  = threadIdx.x;
    const int wave = tid >> 5;
    const int lane = tid & 31;
    const int m    = lane & 15;
    const int half = lane >> 4;

    if (tid < E_DIM) s_bias[tid] = bias[tid];

    const int t0 = blockIdx.x * TOK_PER_BLOCK + wave * 16;
    // A-fragment base: row (t0+m), K chunk offset half*8; second chunk at +16.
    const float* xrow = X + (size_t)(t0 + m) * D_DIM + half * 8;

    v8f acc[ET_TILES];
#pragma unroll
    for (int et = 0; et < ET_TILES; ++et) acc[et] = (v8f){};

    // Prime the pipeline: stage chunk 0 into buffer 0.
    async_copy_chunk(BH, s_bh[0], tid);
    async_copy_chunk(BL, s_bl[0], tid);

    for (int c = 0; c < N_CHUNKS; ++c) {
        const int buf = c & 1;
        if (c + 1 < N_CHUNKS) {
            // Stream the next chunk while this one is consumed.
            async_copy_chunk(BH + (size_t)(c + 1) * CHUNK_ELEMS, s_bh[buf ^ 1], tid);
            async_copy_chunk(BL + (size_t)(c + 1) * CHUNK_ELEMS, s_bl[buf ^ 1], tid);
            // In-order completion: <= 2*UNITS_PER_THREAD outstanding means the
            // current chunk's copies (issued first) have all landed in LDS.
            asm volatile("s_wait_asynccnt %0" :: "i"(2 * UNITS_PER_THREAD) : "memory");
        } else {
            asm volatile("s_wait_asynccnt 0" ::: "memory");
        }
        __syncthreads();

        for (int ktl = 0; ktl < CHUNK_KT; ++ktl) {
            const int kt = c * CHUNK_KT + ktl;
            const float* xp = xrow + kt * 32;
            // stream upcoming X tiles through the prefetcher (speculative)
            __builtin_prefetch(xp + 8 * 32, 0, 0);

            const float4 c0 = *(const float4*)(xp + 0);
            const float4 c1 = *(const float4*)(xp + 4);
            const float4 c2 = *(const float4*)(xp + 16);
            const float4 c3 = *(const float4*)(xp + 20);
            float xv[16] = {c0.x, c0.y, c0.z, c0.w,  c1.x, c1.y, c1.z, c1.w,
                            c2.x, c2.y, c2.z, c2.w,  c3.x, c3.y, c3.z, c3.w};
            v16bf a_hi, a_lo;
#pragma unroll
            for (int i = 0; i < 16; ++i) {
                __bf16 h = (__bf16)xv[i];
                a_hi[i] = h;
                a_lo[i] = (__bf16)(xv[i] - (float)h);
            }

            const __bf16* bhp = &s_bh[buf][(ktl * ET_TILES) * FRAG_ELEMS + lane * 16];
            const __bf16* blp = &s_bl[buf][(ktl * ET_TILES) * FRAG_ELEMS + lane * 16];
#pragma unroll
            for (int et = 0; et < ET_TILES; ++et) {
                v16bf b_hi = *(const v16bf*)(bhp + et * FRAG_ELEMS);
                v16bf b_lo = *(const v16bf*)(blp + et * FRAG_ELEMS);
                // hi*hi + hi*lo + lo*hi  ~= full f32 product, f32 accumulate
                acc[et] = __builtin_amdgcn_wmma_f32_16x16x32_bf16(
                    false, a_hi, false, b_hi, (short)0, acc[et], false, false);
                acc[et] = __builtin_amdgcn_wmma_f32_16x16x32_bf16(
                    false, a_hi, false, b_lo, (short)0, acc[et], false, false);
                acc[et] = __builtin_amdgcn_wmma_f32_16x16x32_bf16(
                    false, a_lo, false, b_hi, (short)0, acc[et], false, false);
            }
        }
        __syncthreads();   // all waves done with buf before it is refilled
    }

    // C layout: VGPR r, lanes 0-15 -> M=r, lanes 16-31 -> M=r+8; N = lane%16.
    const int col = lane & 15;
#pragma unroll
    for (int et = 0; et < ET_TILES; ++et) {
#pragma unroll
        for (int r = 0; r < 8; ++r) {
            int row = wave * 16 + r + (half ? 8 : 0);
            float logit = acc[et][r];
            s_scores[row][et * 16 + col] = 1.0f / (1.0f + __expf(-logit));
        }
    }
    __syncthreads();

    // ---- per-token biased top-8 (threads 0..63) ----
    if (tid < TOK_PER_BLOCK) {
        const int t = blockIdx.x * TOK_PER_BLOCK + tid;
        unsigned long long picked = 0ull;
        float wsel[TOPK];
        int   isel[TOPK];
        float sum = 0.0f;
#pragma unroll
        for (int j = 0; j < TOPK; ++j) {
            float best = -INFINITY;
            int   bi   = 0;
            for (int e = 0; e < E_DIM; ++e) {
                if ((picked >> e) & 1ull) continue;
                float r = s_scores[tid][e] + s_bias[e]; // bias for selection only
                if (r > best) { best = r; bi = e; }     // strict > : first-index tie-break
            }
            picked |= 1ull << bi;
            float w = s_scores[tid][bi];                // gather un-biased score
            wsel[j] = w;
            isel[j] = bi;
            sum += w;
        }
        const float inv = ROUTE_SCALE / (sum + 1e-8f);
#pragma unroll
        for (int j = 0; j < TOPK; ++j) {
            out_w[(size_t)t * TOPK + j] = wsel[j] * inv;
            out_i[(size_t)t * TOPK + j] = isel[j];
        }
    }
}

extern "C" void kernel_launch(void* const* d_in, const int* in_sizes, int n_in,
                              void* d_out, int out_size, void* d_ws, size_t ws_size,
                              hipStream_t stream) {
    const float* X    = (const float*)d_in[0];   // (N, 2048) f32
    const float* W    = (const float*)d_in[1];   // (64, 2048) f32
    const float* bias = (const float*)d_in[2];   // (64,) f32

    const int Ntok = in_sizes[0] / D_DIM;        // 16384

    __bf16* BH = (__bf16*)d_ws;                  // 256 KB
    __bf16* BL = BH + B_TOTAL;                   // 256 KB

    // outputs concatenated: weights (N*8 f32) then indices (N*8 i32)
    float* out_w = (float*)d_out;
    int*   out_i = (int*)d_out + (size_t)Ntok * TOPK;

    prep_w_kernel<<<(B_FRAGS * 32 + 255) / 256, 256, 0, stream>>>(W, BH, BL);
    expert_gate_kernel<<<Ntok / TOK_PER_BLOCK, NTHREADS, 0, stream>>>(
        X, BH, BL, bias, out_w, out_i);
}
